// GroupedQueryAttention_9165460210164
// MI455X (gfx1250) — compile-verified, matrix operand order checked by reading, not run on hardware
//
#include <hip/hip_runtime.h>
#include <hip/hip_bf16.h>
#include <stdint.h>

typedef __attribute__((ext_vector_type(16))) _Float16     v16h;
typedef __attribute__((ext_vector_type(8)))  float        v8f;
typedef __attribute__((ext_vector_type(4)))  unsigned int v4u;
typedef __attribute__((ext_vector_type(8)))  unsigned int v8u;

namespace cfg {
constexpr int B = 2, T = 1024, DIN = 2048;
constexpr int H = 32, HKV = 8, HD = 128;
constexpr int LPREV = 1024, LFULL = 2048;
constexpr int BT = B * T;        // 2048
constexpr int QD = H * HD;       // 4096
constexpr int KD = HKV * HD;     // 1024
constexpr float EPS = 1e-6f;
constexpr float NEGBIG = -1e9f;
constexpr int MB = 128;          // GEMM block M tile (TDM A-panel rows)
constexpr int NB = 256;          // GEMM block N tile (8 waves x 64, 2x4)
}

// ---------------------------------------------------------------- helpers
__device__ __forceinline__ v8f vzero8() {
  v8f v;
#pragma unroll
  for (int i = 0; i < 8; ++i) v[i] = 0.0f;
  return v;
}

// Load a 16x16x32 f16 WMMA A/B operand. Per ISA: lanes 0-15 hold K elements
// {0..7} in v[0..7] and {16..23} in v[8..15]; lanes 16-31 hold {8..15} and
// {24..31}. `rowp` points at the start of this lane's (row, k0) data.
__device__ __forceinline__ v16h load_op16(const _Float16* rowp, int hi) {
  v16h r;
#pragma unroll
  for (int i = 0; i < 8; ++i) {
    r[i]     = rowp[hi * 8 + i];
    r[i + 8] = rowp[16 + hi * 8 + i];
  }
  return r;
}

__device__ __forceinline__ float half16_max(float v) {
#pragma unroll
  for (int m = 1; m <= 8; m <<= 1) v = fmaxf(v, __shfl_xor(v, m, 32));
  return v;
}
__device__ __forceinline__ float half16_sum(float v) {
#pragma unroll
  for (int m = 1; m <= 8; m <<= 1) v += __shfl_xor(v, m, 32);
  return v;
}

// Issue one TDM transfer: MB x 32 f16 panel (rows of stride K) -> LDS, packed
// row-major 32 halves/row. D# group0 = {count, lds_addr, global_addr, type=2};
// group1 = {data_size=2B, tensor dims KxM, tile 32xMB, stride0=K}.
__device__ __forceinline__ void tdm_load_panel(const _Float16* gsrc, unsigned int lds_addr,
                                               int K, int M) {
  const unsigned long long ga = (unsigned long long)(uintptr_t)gsrc;
  v4u g0;
  g0[0] = 1u;                                                    // count=1, user mode
  g0[1] = lds_addr;
  g0[2] = (unsigned int)ga;
  g0[3] = ((unsigned int)(ga >> 32) & 0x01FFFFFFu) | (2u << 30); // addr[56:32] | type=2
  v8u g1;
  g1[0] = 1u << 16;                                              // data_size = 2 bytes
  g1[1] = ((unsigned int)K & 0xFFFFu) << 16;                     // tensor_dim0 lo
  g1[2] = (((unsigned int)K >> 16) & 0xFFFFu) |
          (((unsigned int)M & 0xFFFFu) << 16);                   // dim0 hi | tensor_dim1 lo
  g1[3] = (((unsigned int)M >> 16) & 0xFFFFu) | (32u << 16);     // dim1 hi | tile_dim0=32
  g1[4] = (unsigned int)cfg::MB;                                 // tile_dim1 = MB rows
  g1[5] = (unsigned int)K;                                       // tensor_dim0_stride
  g1[6] = 0u;
  g1[7] = 0u;
  asm volatile("tensor_load_to_lds %0, %1" : : "s"(g0), "s"(g1) : "memory");
}

// ---------------------------------------------------------------- f32 -> f16
__global__ void cvt_kernel(const float* __restrict__ in, _Float16* __restrict__ out, int n) {
  int i = blockIdx.x * blockDim.x + threadIdx.x;
  if (i < n) out[i] = (_Float16)in[i];
}

// ---------------------------------------------------------------- WMMA GEMM
// C[M,N] (f32) = A[M,K] (f16, row-major) * B[N,K]^T (f16, rows are K-vectors)
// Block = 8 waves (2 M x 4 N), 128x256 output per block, 64x64 per wave
// (16 WMMA per 32-wide k-step). A-panel (128x32 f16, 8KB) is staged via the
// Tensor Data Mover into a 2-deep LDS ring; the TDM for k-step i+1 is issued
// BEFORE the 16-WMMA compute of k-step i and waited only afterwards, so DMA
// overlaps matrix math. B operands stream from global per wave.
__global__ __launch_bounds__(256) void gemm_kernel(const _Float16* __restrict__ A,
                                                   const _Float16* __restrict__ Bw,
                                                   float* __restrict__ C,
                                                   int M, int N, int K) {
  using namespace cfg;
  __shared__ __align__(16) _Float16 atile[2][MB * 32];
  const int lane = threadIdx.x & 31;
  const int wave = threadIdx.x >> 5;
  const int hi = lane >> 4, l15 = lane & 15;
  const int wm = wave >> 2;            // 0..1
  const int wn = wave & 3;             // 0..3
  const int m0 = blockIdx.x * MB;
  const int n0 = blockIdx.y * NB;
  const int mloc = wm * 64;            // wave's rows within the A panel
  const int nw = n0 + wn * 64;         // wave's first output column

  v8f acc[4][4];
#pragma unroll
  for (int im = 0; im < 4; ++im)
#pragma unroll
    for (int jn = 0; jn < 4; ++jn) acc[im][jn] = vzero8();

  // prime buffer 0
  if (wave == 0) {
    tdm_load_panel(A + (size_t)m0 * K, (unsigned int)(uintptr_t)&atile[0][0], K, M);
    __builtin_amdgcn_s_wait_tensorcnt(0);
  }
  __syncthreads();

  int cur = 0;
  for (int k0 = 0; k0 < K; k0 += 32) {
    const int nxt = cur ^ 1;
    const bool more = (k0 + 32) < K;
    if (wave == 0 && more)             // async: overlap next panel DMA with compute
      tdm_load_panel(A + (size_t)m0 * K + (k0 + 32),
                     (unsigned int)(uintptr_t)&atile[nxt][0], K, M);

    v16h a[4];
#pragma unroll
    for (int im = 0; im < 4; ++im)
      a[im] = load_op16(&atile[cur][(mloc + im * 16 + l15) * 32], hi);

    if (more) __builtin_prefetch(Bw + (size_t)(nw + l15) * K + k0 + 32, 0, 0);

#pragma unroll
    for (int jn = 0; jn < 4; ++jn) {
      v16h b = load_op16(Bw + (size_t)(nw + jn * 16 + l15) * K + k0, hi);
#pragma unroll
      for (int im = 0; im < 4; ++im)
        acc[im][jn] = __builtin_amdgcn_wmma_f32_16x16x32_f16(false, a[im], false, b,
                                                             (short)0, acc[im][jn],
                                                             false, false);
    }

    if (wave == 0 && more) __builtin_amdgcn_s_wait_tensorcnt(0);
    __syncthreads();
    cur = nxt;
  }

  // C/D layout: VGPR r -> row (r + hi*8), lane&15 -> col
#pragma unroll
  for (int im = 0; im < 4; ++im)
#pragma unroll
    for (int jn = 0; jn < 4; ++jn)
#pragma unroll
      for (int r = 0; r < 8; ++r)
        C[(size_t)(m0 + mloc + im * 16 + r + hi * 8) * N + (nw + jn * 16 + l15)] =
            acc[im][jn][r];
}

// ---------------------------------------------------------------- KV cache prev copy
__global__ void copyprev_kernel(const float* __restrict__ pk, const float* __restrict__ pv,
                                float* __restrict__ fk, float* __restrict__ fv,
                                _Float16* __restrict__ fkh, _Float16* __restrict__ fvh) {
  using namespace cfg;
  const size_t n = (size_t)B * HKV * LPREV * HD;
  size_t i = (size_t)blockIdx.x * blockDim.x + threadIdx.x;
  if (i >= n) return;
  const size_t hd = i % HD;
  const size_t l = (i / HD) % LPREV;
  const size_t bh = i / ((size_t)HD * LPREV);
  const size_t di = (bh * LFULL + l) * HD + hd;
  float k = pk[i], v = pv[i];
  fk[di] = k; fkh[di] = (_Float16)k;
  fv[di] = v; fvh[di] = (_Float16)v;
}

// ---------------------------------------------------------------- RMSNorm + RoPE
__global__ __launch_bounds__(128) void normrope_kernel(
    const float* __restrict__ qf, const float* __restrict__ kf, const float* __restrict__ vf,
    const float* __restrict__ qw, const float* __restrict__ kw,
    const float* __restrict__ cosb, const float* __restrict__ sinb,
    _Float16* __restrict__ qh, float* __restrict__ fullk, float* __restrict__ fullv,
    _Float16* __restrict__ fullkh, _Float16* __restrict__ fullvh) {
  using namespace cfg;
  __shared__ float red[128];
  __shared__ float buf[128];
  const int bt = blockIdx.x;
  const int b = bt / T, t = bt % T;
  const int tid = threadIdx.x;           // == head-dim index
  const int pos = LPREV + t;
  const float c = cosb[(size_t)pos * HD + tid];
  const float s = sinb[(size_t)pos * HD + tid];
  const float qscale = 0.088388347648318447f;  // 1/sqrt(HD) folded into Q

  for (int hh = 0; hh < H; ++hh) {
    float x = qf[(size_t)bt * QD + hh * HD + tid];
    red[tid] = x * x;
    __syncthreads();
    for (int off = 64; off > 0; off >>= 1) {
      if (tid < off) red[tid] += red[tid + off];
      __syncthreads();
    }
    float rr = rsqrtf(red[0] * (1.0f / HD) + EPS);
    float xn = x * rr * qw[tid];
    buf[tid] = xn;
    __syncthreads();
    float rot = (tid < 64) ? -buf[tid + 64] : buf[tid - 64];
    float val = (xn * c + rot * s) * qscale;
    qh[(((size_t)b * H + hh) * T + t) * HD + tid] = (_Float16)val;
    __syncthreads();
  }
  for (int hh = 0; hh < HKV; ++hh) {
    float x = kf[(size_t)bt * KD + hh * HD + tid];
    red[tid] = x * x;
    __syncthreads();
    for (int off = 64; off > 0; off >>= 1) {
      if (tid < off) red[tid] += red[tid + off];
      __syncthreads();
    }
    float rr = rsqrtf(red[0] * (1.0f / HD) + EPS);
    float xn = x * rr * kw[tid];
    buf[tid] = xn;
    __syncthreads();
    float rot = (tid < 64) ? -buf[tid + 64] : buf[tid - 64];
    float val = xn * c + rot * s;
    const size_t di = (((size_t)b * HKV + hh) * LFULL + pos) * HD + tid;
    fullk[di] = val;  fullkh[di] = (_Float16)val;
    float vv = vf[(size_t)bt * KD + hh * HD + tid];
    fullv[di] = vv;   fullvh[di] = (_Float16)vv;
    __syncthreads();
  }
}

// ---------------------------------------------------------------- flash attention
// One wave per 16-query tile of one (b,h). 32 keys per step: 8 WMMA for QK^T,
// online softmax in f32 (row stats replicated across 16-lane halves),
// P transposed C->A layout through per-wave LDS, V tile staged in LDS,
// 8 WMMA for PV. Causal mask applied structurally; early exit past diagonal.
__global__ __launch_bounds__(128) void attn_kernel(const _Float16* __restrict__ qh,
                                                   const _Float16* __restrict__ kh,
                                                   const _Float16* __restrict__ vh,
                                                   _Float16* __restrict__ ctxh) {
  using namespace cfg;
  __shared__ __align__(16) _Float16 pbuf[4][16 * 32];
  __shared__ __align__(16) _Float16 vtile[4][32 * HD];
  const int lane = threadIdx.x & 31;
  const int wave = threadIdx.x >> 5;
  const int hi = lane >> 4, l15 = lane & 15;
  const int tile = blockIdx.x * 4 + wave;
  const int qt = tile & (T / 16 - 1);
  const int h = (tile / (T / 16)) % H;
  const int b = tile / (T / 16) / H;
  const int hkv = h / (H / HKV);

  const _Float16* qbase = qh + (((size_t)b * H + h) * T + (size_t)qt * 16) * HD;
  const _Float16* kbase = kh + ((size_t)b * HKV + hkv) * (size_t)LFULL * HD;
  const _Float16* vbase = vh + ((size_t)b * HKV + hkv) * (size_t)LFULL * HD;
  _Float16* pw = &pbuf[wave][0];
  _Float16* vw = &vtile[wave][0];

  v16h qa[4];
#pragma unroll
  for (int c = 0; c < 4; ++c) qa[c] = load_op16(qbase + (size_t)l15 * HD + c * 32, hi);

  v8f ctx[8];
#pragma unroll
  for (int cn = 0; cn < 8; ++cn) ctx[cn] = vzero8();
  float mrow[8], lrow[8], alpha[8];
#pragma unroll
  for (int r = 0; r < 8; ++r) { mrow[r] = -1e30f; lrow[r] = 0.0f; }

  const int qmax = LPREV + qt * 16 + 15;
  for (int kt = 0; kt < LFULL; kt += 32) {
    if (kt > qmax) break;
    v8f s0 = vzero8(), s1 = vzero8();
#pragma unroll
    for (int c = 0; c < 4; ++c) {   // accumulate over HD in chunks of 32
      v16h bk0 = load_op16(kbase + (size_t)(kt + l15) * HD + c * 32, hi);
      s0 = __builtin_amdgcn_wmma_f32_16x16x32_f16(false, qa[c], false, bk0,
                                                  (short)0, s0, false, false);
      v16h bk1 = load_op16(kbase + (size_t)(kt + 16 + l15) * HD + c * 32, hi);
      s1 = __builtin_amdgcn_wmma_f32_16x16x32_f16(false, qa[c], false, bk1,
                                                  (short)0, s1, false, false);
    }
    // stage V tile (32 keys x HD) into LDS, one key row per lane
    {
      const uint4* src = (const uint4*)(vbase + (size_t)(kt + lane) * HD);
      uint4* dst = (uint4*)(vw + lane * HD);
#pragma unroll
      for (int i = 0; i < (HD * 2) / 16; ++i) dst[i] = src[i];
    }
    // online softmax (rows live in VGPR slots, cols across 16-lane halves)
#pragma unroll
    for (int r = 0; r < 8; ++r) {
      const int qpos = LPREV + qt * 16 + r + hi * 8;
      float x0 = s0[r] + (((kt + l15) <= qpos) ? 0.0f : NEGBIG);
      float x1 = s1[r] + (((kt + 16 + l15) <= qpos) ? 0.0f : NEGBIG);
      float mx = half16_max(fmaxf(x0, x1));
      float mnew = fmaxf(mrow[r], mx);
      alpha[r] = __expf(mrow[r] - mnew);
      mrow[r] = mnew;
      float p0 = __expf(x0 - mnew);
      float p1 = __expf(x1 - mnew);
      lrow[r] = lrow[r] * alpha[r] + half16_sum(p0 + p1);
      pw[(r + hi * 8) * 32 + l15]      = (_Float16)p0;
      pw[(r + hi * 8) * 32 + 16 + l15] = (_Float16)p1;
    }
    asm volatile("s_wait_dscnt 0" ::: "memory");  // LDS writes visible within wave
#pragma unroll
    for (int cn = 0; cn < 8; ++cn)
#pragma unroll
      for (int r = 0; r < 8; ++r) ctx[cn][r] *= alpha[r];
    // P (16x32, A layout from LDS) x V (32x16 per hd chunk, B layout from LDS)
    v16h pa = load_op16(pw + l15 * 32, hi);
#pragma unroll
    for (int cn = 0; cn < 8; ++cn) {
      v16h vb;
#pragma unroll
      for (int i = 0; i < 8; ++i) {
        vb[i]     = vw[(hi * 8 + i) * HD + cn * 16 + l15];
        vb[i + 8] = vw[(16 + hi * 8 + i) * HD + cn * 16 + l15];
      }
      ctx[cn] = __builtin_amdgcn_wmma_f32_16x16x32_f16(false, pa, false, vb,
                                                       (short)0, ctx[cn], false, false);
    }
  }
  // normalize + write ctx as f16 [B, T, H*HD] for the output projection
#pragma unroll
  for (int r = 0; r < 8; ++r) {
    const float inv = 1.0f / lrow[r];
    const int row = r + hi * 8;
    const size_t tg = (size_t)b * T + qt * 16 + row;
#pragma unroll
    for (int cn = 0; cn < 8; ++cn)
      ctxh[tg * QD + (size_t)h * HD + cn * 16 + l15] = (_Float16)(ctx[cn][r] * inv);
  }
}

// ---------------------------------------------------------------- launcher
extern "C" void kernel_launch(void* const* d_in, const int* in_sizes, int n_in,
                              void* d_out, int out_size, void* d_ws, size_t ws_size,
                              hipStream_t stream) {
  using namespace cfg;
  (void)in_sizes; (void)n_in; (void)out_size; (void)ws_size;
  const float* x     = (const float*)d_in[0];
  // d_in[1] attn_mask and d_in[4] position_ids are applied structurally
  const float* cosb  = (const float*)d_in[2];
  const float* sinb  = (const float*)d_in[3];
  const float* prevk = (const float*)d_in[5];
  const float* prevv = (const float*)d_in[6];
  const float* Wq    = (const float*)d_in[7];
  const float* Wk    = (const float*)d_in[8];
  const float* Wv    = (const float*)d_in[9];
  const float* Wo    = (const float*)d_in[10];
  const float* qnw   = (const float*)d_in[11];
  const float* knw   = (const float*)d_in[12];

  float* out   = (float*)d_out;                               // [B,T,DIN]
  float* fullk = out + (size_t)BT * DIN;                      // [B,HKV,LFULL,HD]
  float* fullv = fullk + (size_t)B * HKV * LFULL * HD;

  char* w = (char*)d_ws;
  auto alloc = [&](size_t bytes) -> char* {
    char* p = w;
    w += (bytes + 255) & ~(size_t)255;
    return p;
  };
  _Float16* xh   = (_Float16*)alloc((size_t)BT * DIN * 2);
  _Float16* wqh  = (_Float16*)alloc((size_t)QD * DIN * 2);
  _Float16* wkh  = (_Float16*)alloc((size_t)KD * DIN * 2);
  _Float16* wvh  = (_Float16*)alloc((size_t)KD * DIN * 2);
  _Float16* woh  = (_Float16*)alloc((size_t)DIN * QD * 2);
  float*    qf   = (float*)alloc((size_t)BT * QD * 4);
  float*    kf   = (float*)alloc((size_t)BT * KD * 4);
  float*    vf   = (float*)alloc((size_t)BT * KD * 4);
  _Float16* qhh  = (_Float16*)alloc((size_t)B * H * T * HD * 2);
  _Float16* fkh  = (_Float16*)alloc((size_t)B * HKV * LFULL * HD * 2);
  _Float16* fvh  = (_Float16*)alloc((size_t)B * HKV * LFULL * HD * 2);
  _Float16* ctxh = (_Float16*)alloc((size_t)BT * QD * 2);

  auto cvt = [&](const float* src, _Float16* dst, size_t n) {
    cvt_kernel<<<dim3((unsigned)((n + 255) / 256)), dim3(256), 0, stream>>>(src, dst, (int)n);
  };
  cvt(x,  xh,  (size_t)BT * DIN);
  cvt(Wq, wqh, (size_t)QD * DIN);
  cvt(Wk, wkh, (size_t)KD * DIN);
  cvt(Wv, wvh, (size_t)KD * DIN);
  cvt(Wo, woh, (size_t)DIN * QD);

  gemm_kernel<<<dim3(BT / MB, QD / NB), dim3(256), 0, stream>>>(xh, wqh, qf, BT, QD, DIN);
  gemm_kernel<<<dim3(BT / MB, KD / NB), dim3(256), 0, stream>>>(xh, wkh, kf, BT, KD, DIN);
  gemm_kernel<<<dim3(BT / MB, KD / NB), dim3(256), 0, stream>>>(xh, wvh, vf, BT, KD, DIN);

  {
    size_t n = (size_t)B * HKV * LPREV * HD;
    copyprev_kernel<<<dim3((unsigned)((n + 255) / 256)), dim3(256), 0, stream>>>(
        prevk, prevv, fullk, fullv, fkh, fvh);
  }
  normrope_kernel<<<dim3(BT), dim3(128), 0, stream>>>(qf, kf, vf, qnw, knw, cosb, sinb,
                                                      qhh, fullk, fullv, fkh, fvh);
  attn_kernel<<<dim3(B * H * (T / 16) / 4), dim3(128), 0, stream>>>(qhh, fkh, fvh, ctxh);
  gemm_kernel<<<dim3(BT / MB, DIN / NB), dim3(256), 0, stream>>>(ctxh, woh, out, BT, DIN, QD);
}
